// PlasticLNNXL_62629213110416
// MI455X (gfx1250) — compile-verified, matrix operand order checked by reading, not run on hardware
//
#include <hip/hip_runtime.h>
#include <hip/hip_bf16.h>
#include <math.h>

typedef __attribute__((ext_vector_type(16))) _Float16 v16h;
typedef __attribute__((ext_vector_type(8)))  _Float16 h8;
typedef __attribute__((ext_vector_type(4)))  _Float16 h4;
typedef __attribute__((ext_vector_type(8)))  float    v8f;

#define WMMA_F32_F16(a, b, c) \
  __builtin_amdgcn_wmma_f32_16x16x32_f16(false, (a), false, (b), (short)0, (c), false, false)

__device__ inline float gelu_f(float x)    { return 0.5f * x * (1.f + erff(x * 0.70710678118654752f)); }
__device__ inline float sigmoid_f(float x) { return 1.f / (1.f + expf(-x)); }

// concat two 8-half runs into one 16-half WMMA operand (no per-element moves)
__device__ inline v16h cat16(h8 lo, h8 hi) {
  return __builtin_shufflevector(lo, hi, 0, 1, 2, 3, 4, 5, 6, 7, 8, 9, 10, 11, 12, 13, 14, 15);
}

// load 8 contiguous f32, scale, pack to 8 halves (two dwordx4 loads + cvt)
__device__ inline h8 pack8(const float* __restrict__ p, float s) {
  float4 a = *(const float4*)p;
  float4 b = *(const float4*)(p + 4);
  h8 r;
  r[0] = (_Float16)(a.x * s); r[1] = (_Float16)(a.y * s);
  r[2] = (_Float16)(a.z * s); r[3] = (_Float16)(a.w * s);
  r[4] = (_Float16)(b.x * s); r[5] = (_Float16)(b.y * s);
  r[6] = (_Float16)(b.z * s); r[7] = (_Float16)(b.w * s);
  return r;
}

// ---------------------------------------------------------------------------
// WMMA GEMM: C[M,N] = act(A[M,K] @ B(K,N) + bias) (+= if addC)
// A row-major. B row-major (K,N) or, if transB, (N,K) row-major.
// Block tile 128x64, BK=32. 8 waves as 4(M) x 2(N); wave tile 32x32 =
// 2x2 WMMA accumulators, each A/B fragment reused twice.
// LDS tiles stored fragment-major so every fragment = two ds_load_b128.
// ---------------------------------------------------------------------------
#define BM 128
#define BN 64
#define BKK 32

__global__ __launch_bounds__(256)
void gemm_wmma_f16(const float* __restrict__ A, const float* __restrict__ B,
                   const float* __restrict__ bias, float* __restrict__ C,
                   int M, int N, int K, int transB, int addC, int act)
{
  __shared__ _Float16 As[BM][BKK];    // [m][k], 8 KB
  __shared__ _Float16 Bst[BN][BKK];   // [n][k] (k-major), 4 KB

  const int bm = blockIdx.y * BM;
  const int bn = blockIdx.x * BN;
  const int tid  = threadIdx.x;
  const int wave = tid >> 5;
  const int lane = tid & 31;
  const int wm = wave >> 1;            // 0..3
  const int wn = wave & 1;             // 0..1
  const int rows0 = wm * 32;
  const int col0  = wn * 32;
  const int grp = lane >> 4;           // half-wave group
  const int l   = lane & 15;
  const int kb  = grp * 8;             // per-ISA 16-bit A/B layout K sub-base

  // staging coordinates
  const int ar  = tid >> 1;            // A row 0..127
  const int acb = (tid & 1) * 16;      // A col base 0/16 (16 elems/thread)
  const int bnn = tid >> 2;            // B n 0..63
  const int bks = (tid & 3) * 8;       // B k base 0,8,16,24 (8 elems/thread)

  v8f zero = {};
  v8f acc[2][2];
  acc[0][0] = zero; acc[0][1] = zero; acc[1][0] = zero; acc[1][1] = zero;

  for (int k0 = 0; k0 < K; k0 += BKK) {
    if (k0 + BKK < K)
      __builtin_prefetch(&A[(long long)bm * K + k0 + BKK], 0, 1);

    // ---- stage A tile (fp32 -> fp16), vectorized fast path ----
    {
      int gr = bm + ar;
      h8 p0, p1;
      if (gr < M && (k0 + acb + 15) < K) {
        const float* ap = A + (long long)gr * K + k0 + acb;
        p0 = pack8(ap, 1.f);
        p1 = pack8(ap + 8, 1.f);
      } else {
#pragma unroll
        for (int j = 0; j < 8; ++j) {
          int gc0 = k0 + acb + j, gc1 = gc0 + 8;
          p0[j] = (_Float16)((gr < M && gc0 < K) ? A[(long long)gr * K + gc0] : 0.f);
          p1[j] = (_Float16)((gr < M && gc1 < K) ? A[(long long)gr * K + gc1] : 0.f);
        }
      }
      *(h8*)&As[ar][acb]     = p0;
      *(h8*)&As[ar][acb + 8] = p1;
    }
    // ---- stage B tile (k-major, optional transpose) ----
    {
      int gn = bn + bnn;
      h8 q;
      if (transB && gn < N && (k0 + bks + 7) < K) {
        q = pack8(B + (long long)gn * K + k0 + bks, 1.f);
      } else {
#pragma unroll
        for (int j = 0; j < 8; ++j) {
          int gk = k0 + bks + j;
          float v = 0.f;
          if (gk < K && gn < N)
            v = transB ? B[(long long)gn * K + gk] : B[(long long)gk * N + gn];
          q[j] = (_Float16)v;
        }
      }
      *(h8*)&Bst[bnn][bks] = q;
    }
    __syncthreads();

    // ---- fragments: two b128 LDS loads each ----
    v16h af[2], bf[2];
#pragma unroll
    for (int i = 0; i < 2; ++i) {
      const int m = rows0 + 16 * i + l;
      af[i] = cat16(*(const h8*)&As[m][kb], *(const h8*)&As[m][kb + 16]);
    }
#pragma unroll
    for (int j = 0; j < 2; ++j) {
      const int n = col0 + 16 * j + l;
      bf[j] = cat16(*(const h8*)&Bst[n][kb], *(const h8*)&Bst[n][kb + 16]);
    }
#pragma unroll
    for (int i = 0; i < 2; ++i)
#pragma unroll
      for (int j = 0; j < 2; ++j)
        acc[i][j] = WMMA_F32_F16(af[i], bf[j], acc[i][j]);
    __syncthreads();
  }

  // ---- epilogue: bias / activation / optional accumulate ----
#pragma unroll
  for (int i = 0; i < 2; ++i) {
#pragma unroll
    for (int j = 0; j < 2; ++j) {
#pragma unroll
      for (int r = 0; r < 8; ++r) {
        int row = bm + rows0 + 16 * i + r + 8 * grp;
        int col = bn + col0 + 16 * j + l;
        if (row < M && col < N) {
          float v = acc[i][j][r] + (bias ? bias[col] : 0.f);
          if (act == 1) v = gelu_f(v); else if (act == 2) v = sigmoid_f(v);
          long long o = (long long)row * N + col;
          C[o] = addC ? (C[o] + v) : v;
        }
      }
    }
  }
}

// ---------------------------------------------------------------------------
// Flash attention: 4 waves / block, 64 queries / block, shared K/V staging.
// 32 keys per iteration: 8 WMMA (Q.K^T) + 8 WMMA (P.V) per wave.
// K staged [key][dim], V staged [dim][key] so all fragments are b128 loads.
// Softmax scale folded into Q fragments.
// ---------------------------------------------------------------------------
__global__ __launch_bounds__(128)
void flash_attn(const float* __restrict__ Q, const float* __restrict__ Kx,
                const float* __restrict__ V, float* __restrict__ O,
                int S, int NH, int HD)
{
  const int h = blockIdx.y, b = blockIdx.z;
  const int tid  = threadIdx.x;
  const int wave = tid >> 5;
  const int lane = tid & 31;
  const int grp = lane >> 4, l = lane & 15, kb = grp * 8;
  const int rs = NH * HD;
  const long long base = (long long)b * S * rs + h * HD;
  const float* Qb = Q + base;
  const float* Kb = Kx + base;
  const float* Vb = V + base;
  float* Ob = O + base;
  const float scale = rsqrtf((float)HD);
  const int m = blockIdx.x * 64 + wave * 16 + l;   // this lane's query row

  __shared__ _Float16 Kt[32][128];     // [key][dim]   8 KB
  __shared__ _Float16 Vt[128][32];     // [dim][key]   8 KB
  __shared__ _Float16 Ps[4][16][32];   // per-wave P   4 KB

  // Q fragments (16x128 -> 4 chunks of K=32), pre-scaled, vectorized loads
  v16h qf[4];
#pragma unroll
  for (int t = 0; t < 4; ++t) {
    const float* qr = Qb + (long long)m * rs + t * 32;
    qf[t] = cat16(pack8(qr + kb, scale), pack8(qr + kb + 16, scale));
  }

  v8f zero = {};
  v8f o[8];
#pragma unroll
  for (int t = 0; t < 8; ++t) o[t] = zero;
  float mrow[8], lrow[8];
#pragma unroll
  for (int r = 0; r < 8; ++r) { mrow[r] = -3.0e38f; lrow[r] = 0.f; }

  for (int j0 = 0; j0 < S; j0 += 32) {
    // ---- cooperative K/V staging: coalesced float4 loads ----
#pragma unroll
    for (int it = 0; it < 8; ++it) {
      int g  = tid + it * 128;          // 0..1023 float4-groups
      int ky = g >> 5;                  // key 0..31
      int dq = (g & 31) * 4;            // dim 0..124
      float4 kv = *(const float4*)&Kb[(long long)(j0 + ky) * rs + dq];
      h4 pk;
      pk[0] = (_Float16)kv.x; pk[1] = (_Float16)kv.y;
      pk[2] = (_Float16)kv.z; pk[3] = (_Float16)kv.w;
      *(h4*)&Kt[ky][dq] = pk;
      float4 vv = *(const float4*)&Vb[(long long)(j0 + ky) * rs + dq];
      Vt[dq + 0][ky] = (_Float16)vv.x;
      Vt[dq + 1][ky] = (_Float16)vv.y;
      Vt[dq + 2][ky] = (_Float16)vv.z;
      Vt[dq + 3][ky] = (_Float16)vv.w;
    }
    __syncthreads();

    // ---- scores: 16x32 (two 16-key tiles) ----
    v8f s0a = zero, s1a = zero;
#pragma unroll
    for (int t = 0; t < 4; ++t) {
      v16h bf = cat16(*(const h8*)&Kt[l][t * 32 + kb],
                      *(const h8*)&Kt[l][t * 32 + kb + 16]);
      s0a = WMMA_F32_F16(qf[t], bf, s0a);
    }
#pragma unroll
    for (int t = 0; t < 4; ++t) {
      v16h bf = cat16(*(const h8*)&Kt[16 + l][t * 32 + kb],
                      *(const h8*)&Kt[16 + l][t * 32 + kb + 16]);
      s1a = WMMA_F32_F16(qf[t], bf, s1a);
    }

    // ---- online softmax (rows live across each 16-lane group) ----
    float alpha[8], p0v[8], p1v[8];
#pragma unroll
    for (int r = 0; r < 8; ++r) {
      float s0 = s0a[r], s1 = s1a[r];
      float mx = fmaxf(s0, s1);
#pragma unroll
      for (int off = 8; off > 0; off >>= 1) mx = fmaxf(mx, __shfl_xor(mx, off, 16));
      float mnew = fmaxf(mrow[r], mx);
      alpha[r] = expf(mrow[r] - mnew);
      float p0 = expf(s0 - mnew), p1 = expf(s1 - mnew);
      float ls = p0 + p1;
#pragma unroll
      for (int off = 8; off > 0; off >>= 1) ls += __shfl_xor(ls, off, 16);
      lrow[r] = lrow[r] * alpha[r] + ls;
      mrow[r] = mnew;
      p0v[r] = p0; p1v[r] = p1;
    }

    // ---- P: C-layout -> A-layout through per-wave LDS tile ----
#pragma unroll
    for (int r = 0; r < 8; ++r) {
      Ps[wave][r + 8 * grp][l]      = (_Float16)p0v[r];
      Ps[wave][r + 8 * grp][16 + l] = (_Float16)p1v[r];
    }
    __syncthreads();
    v16h pf = cat16(*(const h8*)&Ps[wave][l][kb],
                    *(const h8*)&Ps[wave][l][kb + 16]);

    // ---- O = diag(alpha) O + P V ----
#pragma unroll
    for (int t = 0; t < 8; ++t) {
      v16h vf = cat16(*(const h8*)&Vt[16 * t + l][kb],
                      *(const h8*)&Vt[16 * t + l][kb + 16]);
      v8f on = o[t];
#pragma unroll
      for (int r = 0; r < 8; ++r) on[r] *= alpha[r];
      o[t] = WMMA_F32_F16(pf, vf, on);
    }
    __syncthreads();   // protect Kt/Vt before next staging pass
  }

  // ---- finalize: divide by row sums ----
#pragma unroll
  for (int t = 0; t < 8; ++t)
#pragma unroll
    for (int r = 0; r < 8; ++r) {
      int row = blockIdx.x * 64 + wave * 16 + r + 8 * grp;
      Ob[(long long)row * rs + t * 16 + l] = o[t][r] / lrow[r];
    }
}

// ---------------------------------------------------------------------------
// LayerNorm over D (=2048), one block per row; optional broadcast add vector
// indexed (b, d) with b = row / S. In-place safe (row cached in registers).
// ---------------------------------------------------------------------------
__global__ __launch_bounds__(256)
void ln_rows(const float* __restrict__ in, const float* __restrict__ addvec,
             const float* __restrict__ g, const float* __restrict__ beta,
             float* __restrict__ out, int S, int D)
{
  const int row = blockIdx.x;
  const int b = row / S;
  const float* xr = in + (long long)row * D;
  float* orow = out + (long long)row * D;
  __shared__ float sred[256];
  const int tid = threadIdx.x;
  const int nIt = D >> 8;          // D/256 (<= 8)
  float loc[8];
  float sum = 0.f;
  for (int i = 0; i < nIt; ++i) {
    int c = tid + (i << 8);
    float v = xr[c];
    if (addvec) v += addvec[(long long)b * D + c];
    loc[i] = v; sum += v;
  }
  sred[tid] = sum; __syncthreads();
  for (int s2 = 128; s2 > 0; s2 >>= 1) { if (tid < s2) sred[tid] += sred[tid + s2]; __syncthreads(); }
  float mean = sred[0] / D; __syncthreads();
  float var = 0.f;
  for (int i = 0; i < nIt; ++i) { float d = loc[i] - mean; var += d * d; }
  sred[tid] = var; __syncthreads();
  for (int s2 = 128; s2 > 0; s2 >>= 1) { if (tid < s2) sred[tid] += sred[tid + s2]; __syncthreads(); }
  float inv = rsqrtf(sred[0] / D + 1e-5f);
  for (int i = 0; i < nIt; ++i) {
    int c = tid + (i << 8);
    orow[c] = (loc[i] - mean) * inv * g[c] + beta[c];
  }
}

// RoPE (concatenated-halves variant, matching the reference), in place.
__global__ __launch_bounds__(64)
void rope_qk(float* __restrict__ q, float* __restrict__ k, int S, int NH, int HD)
{
  const int idx = blockIdx.x;
  const int h = idx % NH;
  const int bs = idx / NH;
  const int s = bs % S;
  const int rs = NH * HD;
  float* qr = q + (long long)bs * rs + h * HD;
  float* kr = k + (long long)bs * rs + h * HD;
  const int j = threadIdx.x;                 // 0..63
  float invf = powf(10000.f, -2.f * (float)j / (float)HD);
  float f = (float)s * invf;
  float c = cosf(f), sn = sinf(f);
  float q1 = qr[2 * j], q2 = qr[2 * j + 1];
  float k1 = kr[2 * j], k2 = kr[2 * j + 1];
  __syncthreads();
  qr[j] = q1 * c - q2 * sn;  qr[j + 64] = q1 * sn + q2 * c;
  kr[j] = k1 * c - k2 * sn;  kr[j + 64] = k1 * sn + k2 * c;
}

// mean over S: out[b*D+d] = mean_s x[b,s,d]
__global__ void mean_over_s(const float* __restrict__ x, float* __restrict__ out,
                            int Bb, int S, int D)
{
  int i = blockIdx.x * blockDim.x + threadIdx.x;
  if (i >= Bb * D) return;
  int b = i / D, d = i % D;
  float s = 0.f;
  for (int t = 0; t < S; ++t) s += x[((long long)b * S + t) * D + d];
  out[i] = s / (float)S;
}

// new_hf = DT * gate * xp / (TAU_MIN + (TAU_MAX-TAU_MIN)*sigmoid(tau_ffn))
__global__ void hf_vec(const float* __restrict__ gate, const float* __restrict__ xp,
                       const float* __restrict__ tau, float* __restrict__ outv,
                       int n, int D)
{
  int i = blockIdx.x * blockDim.x + threadIdx.x;
  if (i >= n) return;
  int d = i % D;
  float tf = 0.1f + 9.9f * sigmoid_f(tau[d]);
  outv[i] = 0.1f * gate[i] * xp[i] / tf;
}

__global__ void add_vec(const float* __restrict__ a, const float* __restrict__ b,
                        float* __restrict__ o, int n)
{
  int i = blockIdx.x * blockDim.x + threadIdx.x;
  if (i < n) o[i] = a[i] + b[i];
}

// row softmax with pre-scale, one block per row
__global__ __launch_bounds__(256)
void softmax_rows(float* __restrict__ x, int N, float scale)
{
  const int row = blockIdx.x;
  float* p = x + (long long)row * N;
  __shared__ float sred[256];
  const int tid = threadIdx.x;
  float mx = -3.0e38f;
  for (int i = tid; i < N; i += 256) mx = fmaxf(mx, p[i] * scale);
  sred[tid] = mx; __syncthreads();
  for (int s2 = 128; s2 > 0; s2 >>= 1) { if (tid < s2) sred[tid] = fmaxf(sred[tid], sred[tid + s2]); __syncthreads(); }
  mx = sred[0]; __syncthreads();
  float sum = 0.f;
  for (int i = tid; i < N; i += 256) { float e = expf(p[i] * scale - mx); p[i] = e; sum += e; }
  sred[tid] = sum; __syncthreads();
  for (int s2 = 128; s2 > 0; s2 >>= 1) { if (tid < s2) sred[tid] += sred[tid + s2]; __syncthreads(); }
  float inv = 1.f / sred[0];
  for (int i = tid; i < N; i += 256) p[i] *= inv;
}

// ---------------------------------------------------------------------------
// host orchestration
// ---------------------------------------------------------------------------
static inline void launch_gemm(const float* A, const float* B, const float* bias, float* C,
                               int M, int N, int K, int transB, int addC, int act,
                               hipStream_t s)
{
  dim3 g((N + BN - 1) / BN, (M + BM - 1) / BM, 1);
  gemm_wmma_f16<<<g, 256, 0, s>>>(A, B, bias, C, M, N, K, transB, addC, act);
}

extern "C" void kernel_launch(void* const* d_in, const int* in_sizes, int n_in,
                              void* d_out, int out_size, void* d_ws, size_t ws_size,
                              hipStream_t stream)
{
  (void)in_sizes; (void)n_in; (void)out_size; (void)ws_size;

  const int Bb = 4, S = 1024, DI = 1024, HID = 2048, L = 2, NH = 16, HD = 128;
  const int T = Bb * S;                    // 4096 tokens
  const int H4 = 4 * HID;                  // 8192
  const int MEM_N = 10000, MEM_D = 512, OUT_D = 256;

  // ---- inputs (setup_inputs order) ----
  const float* x_in   = (const float*)d_in[0];
  const float* w_in   = (const float*)d_in[1];
  const float* b_in   = (const float*)d_in[2];
  const float* g_in   = (const float*)d_in[3];
  const float* be_in  = (const float*)d_in[4];
  const float* wq     = (const float*)d_in[5];
  const float* wk     = (const float*)d_in[6];
  const float* wv     = (const float*)d_in[7];
  const float* wo     = (const float*)d_in[8];
  // d_in[9] tau_base, d_in[10] w_tg: dead code in the reference
  const float* g1     = (const float*)d_in[11];
  const float* be1    = (const float*)d_in[12];
  const float* g2     = (const float*)d_in[13];
  const float* be2    = (const float*)d_in[14];
  const float* w_ff1  = (const float*)d_in[15];
  const float* b_ff1  = (const float*)d_in[16];
  const float* w_ff2  = (const float*)d_in[17];
  const float* b_ff2  = (const float*)d_in[18];
  const float* w_gate = (const float*)d_in[19];
  const float* b_gate = (const float*)d_in[20];
  const float* tau_ffn= (const float*)d_in[21];
  const float* mem_bank=(const float*)d_in[22];
  const float* w_mp   = (const float*)d_in[23];
  const float* b_mp   = (const float*)d_in[24];
  const float* w_mr   = (const float*)d_in[25];
  const float* b_mr   = (const float*)d_in[26];
  const float* w_a1   = (const float*)d_in[27];
  const float* b_a1   = (const float*)d_in[28];
  const float* w_a2   = (const float*)d_in[29];
  const float* b_a2   = (const float*)d_in[30];
  const float* w_v1   = (const float*)d_in[31];
  const float* b_v1   = (const float*)d_in[32];
  const float* w_v2   = (const float*)d_in[33];
  const float* b_v2   = (const float*)d_in[34];

  float* outp = (float*)d_out;             // [action (4x256) | value (4x1)]

  // ---- workspace layout (floats) ----
  float* ws = (float*)d_ws;
  const size_t TOK = (size_t)T * HID;      // 8,388,608
  float* xbuf  = ws;
  float* ybuf  = xbuf + TOK;
  float* region= ybuf + TOK;               // 4*TOK: q|k|v|attn, reused as ffn hidden
  float* qbuf  = region;
  float* kbuf  = region + TOK;
  float* vbuf  = region + 2 * TOK;
  float* abuf  = region + 3 * TOK;
  float* hbuf  = region;                   // (T x 8192) after attention consumed
  float* pooled= region + 4 * TOK;
  float* gatev = pooled + Bb * HID;
  float* addv  = gatev + Bb * HID;
  float* qm    = addv + Bb * HID;
  float* scores= qm + Bb * MEM_D;
  float* memv  = scores + (size_t)Bb * MEM_N;
  float* memp  = memv + Bb * MEM_D;
  float* feat  = memp + Bb * HID;
  float* t1    = feat + Bb * HID;
  float* t2    = t1 + Bb * (HID / 2);

  // ---- x = LN(x_in @ w_in + b_in) ----
  launch_gemm(x_in, w_in, b_in, xbuf, T, HID, DI, 0, 0, 0, stream);
  ln_rows<<<T, 256, 0, stream>>>(xbuf, nullptr, g_in, be_in, xbuf, S, HID);

  for (int i = 0; i < L; ++i) {
    const long long wOff  = (long long)i * HID * HID;
    const long long f1Off = (long long)i * HID * H4;
    const long long f2Off = (long long)i * H4 * HID;

    // q/k/v projections
    launch_gemm(xbuf, wq + wOff, nullptr, qbuf, T, HID, HID, 0, 0, 0, stream);
    launch_gemm(xbuf, wk + wOff, nullptr, kbuf, T, HID, HID, 0, 0, 0, stream);
    launch_gemm(xbuf, wv + wOff, nullptr, vbuf, T, HID, HID, 0, 0, 0, stream);

    // RoPE on q, k
    rope_qk<<<T * NH, 64, 0, stream>>>(qbuf, kbuf, S, NH, HD);

    // attention -> abuf
    {
      dim3 g(S / 64, NH, Bb);
      flash_attn<<<g, 128, 0, stream>>>(qbuf, kbuf, vbuf, abuf, S, NH, HD);
    }

    // x += attn @ wo
    launch_gemm(abuf, wo + wOff, nullptr, xbuf, T, HID, HID, 0, 1, 0, stream);

    // y = LN(x); x += gelu(y@w_ff1+b1) @ w_ff2 + b2
    ln_rows<<<T, 256, 0, stream>>>(xbuf, nullptr, g1 + (size_t)i * HID, be1 + (size_t)i * HID,
                                   ybuf, S, HID);
    launch_gemm(ybuf, w_ff1 + f1Off, b_ff1 + (size_t)i * H4, hbuf, T, H4, HID, 0, 0, 1, stream);
    launch_gemm(hbuf, w_ff2 + f2Off, b_ff2 + (size_t)i * HID, xbuf, T, HID, H4, 0, 1, 0, stream);

    // gate path: xp = mean_s(x); gate = sigmoid(xp@w_gate+b); x = LN(x + new_hf)
    mean_over_s<<<(Bb * HID + 255) / 256, 256, 0, stream>>>(xbuf, pooled, Bb, S, HID);
    launch_gemm(pooled, w_gate + wOff, b_gate + (size_t)i * HID, gatev, Bb, HID, HID, 0, 0, 2, stream);
    hf_vec<<<(Bb * HID + 255) / 256, 256, 0, stream>>>(gatev, pooled, tau_ffn + (size_t)i * HID,
                                                       addv, Bb * HID, HID);
    ln_rows<<<T, 256, 0, stream>>>(xbuf, addv, g2 + (size_t)i * HID, be2 + (size_t)i * HID,
                                   xbuf, S, HID);
  }

  // ---- memory retrieval head ----
  mean_over_s<<<(Bb * HID + 255) / 256, 256, 0, stream>>>(xbuf, pooled, Bb, S, HID);
  launch_gemm(pooled, w_mp, b_mp, qm, Bb, MEM_D, HID, 0, 0, 0, stream);
  launch_gemm(qm, mem_bank, nullptr, scores, Bb, MEM_N, MEM_D, 1, 0, 0, stream);  // qm @ mem^T
  softmax_rows<<<Bb, 256, 0, stream>>>(scores, MEM_N, rsqrtf((float)MEM_D));
  launch_gemm(scores, mem_bank, nullptr, memv, Bb, MEM_D, MEM_N, 0, 0, 0, stream);
  launch_gemm(memv, w_mr, b_mr, memp, Bb, HID, MEM_D, 0, 0, 0, stream);
  add_vec<<<(Bb * HID + 255) / 256, 256, 0, stream>>>(pooled, memp, feat, Bb * HID);

  // ---- action / value heads ----
  launch_gemm(feat, w_a1, b_a1, t1, Bb, HID / 2, HID, 0, 0, 1, stream);
  launch_gemm(t1, w_a2, b_a2, outp, Bb, OUT_D, HID / 2, 0, 0, 0, stream);
  launch_gemm(feat, w_v1, b_v1, t2, Bb, HID / 2, HID, 0, 0, 1, stream);
  launch_gemm(t2, w_v2, b_v2, outp + Bb * OUT_D, Bb, 1, HID / 2, 0, 0, 0, stream);
}
